// Generator_67705864454707
// MI455X (gfx1250) — matrix operand order repaired and checked by reading, not compiled, not run on hardware
//
#include <hip/hip_runtime.h>
#include <math.h>

// ---------------------------------------------------------------------------
// MI455X (gfx1250) implementation.
//  * Heavy GEMMs (filterbank conv 68.7 GF, dilated convs 2.2 GF) on
//    v_wmma_f32_16x16x32_bf16, operands staged in LDS as bf16, f32 accum.
//    sched_group_barrier pipelines each K-step: all DS reads first, then the
//    WMMA burst; launch_bounds(...,1) lifts the VGPR cap (LDS already limits
//    us to 1 WG/WGP, so registers are free).
//  * Bulk global->LDS weight staging via GLOBAL_LOAD_ASYNC_TO_LDS_B128
//    (ASYNCcnt) -- CDNA5 async path, no VGPR round trip.
//  * 65536-pt rffts done as packed complex 32768-pt radix-2 FFTs fully in
//    256KB LDS (CDNA5 WGP has 320KB).  irfft linearity turns 512 event
//    inverse FFTs into 8.
//  * Deterministic: no floating-point atomics anywhere.
// Workspace requirement: ~16.5 MB.
// ---------------------------------------------------------------------------

#define PI_F  3.14159265358979323846f
#define TPI_F 6.28318530717958647692f
#define PI_D  3.14159265358979323846264338

#if __has_builtin(__builtin_amdgcn_sched_group_barrier)
#define SCHED_GROUP(mask, size, id) __builtin_amdgcn_sched_group_barrier(mask, size, id)
#else
#define SCHED_GROUP(mask, size, id)
#endif

typedef __attribute__((ext_vector_type(16))) __bf16 v16bf;
typedef __attribute__((ext_vector_type(8)))  float  v8f;

__device__ __forceinline__ unsigned short f2bf(float f) {
  unsigned u = __float_as_uint(f);
  u = (u + 0x7FFFu + ((u >> 16) & 1u)) >> 16;  // round-to-nearest-even
  return (unsigned short)u;
}
__device__ __forceinline__ __bf16 bfb(unsigned short u) {
  union { unsigned short s; __bf16 h; } c; c.s = u; return c.h;
}
__device__ __forceinline__ v8f wmma_bf16(v16bf a, v16bf b, v8f c) {
  return __builtin_amdgcn_wmma_f32_16x16x32_bf16(false, a, false, b, (short)0, c, false, false);
}
__device__ __forceinline__ float2 cmulf(float2 a, float2 b) {
  return make_float2(a.x*b.x - a.y*b.y, a.x*b.y + a.y*b.x);
}
__device__ __forceinline__ float lrelu(float v) { return v >= 0.f ? v : 0.2f*v; }

// Async 16B global->LDS copy (GLOBAL_LOAD_ASYNC_TO_LDS_B128, ASYNCcnt).
// ldsOff = LDS byte offset (low 32 bits of a generic __shared__ pointer:
// generic->LDS mapping truncates upper bits per CDNA5 ISA 10.2).
__device__ __forceinline__ void async_copy16(unsigned ldsOff, const void* gptr) {
  asm volatile("global_load_async_to_lds_b128 %0, %1, off"
               :: "v"(ldsOff), "v"(gptr) : "memory");
}
__device__ __forceinline__ void wait_async() {
  asm volatile("s_wait_asynccnt 0" ::: "memory");
}

// Build A fragment (16x32 bf16, contiguous k = 16*(lane/16) + s) from row-major LDS ushort.
__device__ __forceinline__ v16bf load_a(const unsigned short* row, int lane) {
  v16bf a;
  int kh = (lane >= 16) ? 16 : 0;
#pragma unroll
  for (int v = 0; v < 8; ++v) {
    int kl = (v << 1) + kh;
    a[2*v]   = bfb(row[kl]);
    a[2*v+1] = bfb(row[kl+1]);
  }
  return a;
}

// ------------------------------- prep --------------------------------------
__global__ void k_cvt(const float* __restrict__ s, unsigned short* __restrict__ d, int n) {
  int i = blockIdx.x*blockDim.x + threadIdx.x;
  if (i < n) d[i] = f2bf(s[i]);
}

__global__ __launch_bounds__(256) void k_atomnorm(const float* __restrict__ atoms,
                                                  float* __restrict__ inv) {
  __shared__ float wred[8];
  int row = blockIdx.x, tid = threadIdx.x;
  const float* a = atoms + (size_t)row*2048;
  float s = 0.f;
  for (int i = tid; i < 2048; i += 256) { float v = a[i]; s += v*v; }
  for (int o = 16; o > 0; o >>= 1) s += __shfl_down(s, o, 32);
  if ((tid & 31) == 0) wred[tid >> 5] = s;
  __syncthreads();
  if (tid == 0) {
    float t = 0.f;
    for (int w = 0; w < 8; ++w) t += wred[w];
    inv[row] = 1.f/(sqrtf(t) + 1e-8f);
  }
}

// -------------------------- filterbank + pooling ---------------------------
// bands[b,c,t] = sum_k fb[c,k] * x[b, t-256+k];  chunk sums of |bands| over
// 128-sample chunks (pooling window 512 stride 256 pad 128 == sum of 4 chunks).
__global__ __launch_bounds__(512, 1) void k_filterbank(const float* __restrict__ x,
    const unsigned short* __restrict__ fbbf, float* __restrict__ schunk) {
  extern __shared__ unsigned char smem[];
  unsigned short* fbl = (unsigned short*)smem;             // 128*512 bf16
  unsigned short* xl  = (unsigned short*)(smem + 131072);  // 768 window
  float* sred = (float*)(smem + 131072 + 2048);            // 2*128
  int tid = threadIdx.x, lane = tid & 31, wave = tid >> 5;
  int b = blockIdx.x >> 7, tile = blockIdx.x & 127;
  int t0 = tile << 8;
  // async-stage the 128KB bf16 filterbank into LDS (8192 x 16B chunks)
  {
    unsigned ldsBase = (unsigned)(unsigned long long)(void*)fbl;
    const char* g = (const char*)fbbf;
    for (int i = tid; i < 8192; i += 512)
      async_copy16(ldsBase + (unsigned)i*16u, g + (size_t)i*16u);
  }
  const float* xb = x + (size_t)b*32768;
  if (tid == 0) __builtin_prefetch(xb + ((t0 + 512) & 32767), 0, 0);
  for (int i = tid; i < 768; i += 512) {
    int t = t0 - 256 + i;
    xl[i] = f2bf((t >= 0 && t < 32768) ? xb[t] : 0.f);
  }
  wait_async();
  __syncthreads();
  int mt = wave >> 1, nh = wave & 1;
  v8f acc[8];
#pragma unroll
  for (int j = 0; j < 8; ++j)
#pragma unroll
    for (int r = 0; r < 8; ++r) acc[j][r] = 0.f;
  int kb = (lane >= 16) ? 16 : 0;
  const unsigned short* arow = fbl + (mt*16 + (lane & 15))*512;
  for (int K0 = 0; K0 < 512; K0 += 32) {
    v16bf a = load_a(arow + K0, lane);
    v16bf bbf[8];
#pragma unroll
    for (int j = 0; j < 8; ++j) {
      int ncol = (nh*8 + j)*16 + (lane & 15);
      const unsigned short* bp = xl + ncol + K0 + kb;   // B[k][n] = x[n + k]
#pragma unroll
      for (int v = 0; v < 8; ++v) { bbf[j][2*v] = bfb(bp[2*v]); bbf[j][2*v+1] = bfb(bp[2*v+1]); }
    }
#pragma unroll
    for (int j = 0; j < 8; ++j) acc[j] = wmma_bf16(a, bbf[j], acc[j]);
    // pipeline hint: cluster the 18 ds reads (2 A + 16 B), then 8 WMMAs
    SCHED_GROUP(0x100, 18, 0);
    SCHED_GROUP(0x008, 8, 0);
  }
  if (tid < 256) sred[tid] = 0.f;
  __syncthreads();
#pragma unroll
  for (int r = 0; r < 8; ++r) {
    float s = 0.f;
#pragma unroll
    for (int j = 0; j < 8; ++j) s += fabsf(acc[j][r]);
    for (int o = 8; o > 0; o >>= 1) s += __shfl_xor(s, o, 32);
    if ((lane & 15) == 0) {
      int ch = mt*16 + ((lane >= 16) ? r + 8 : r);
      sred[nh*128 + ch] = s;                             // single writer
    }
  }
  __syncthreads();
  if (tid < 256) {
    int hh = tid >> 7, ch = tid & 127;
    schunk[((size_t)b*128 + ch)*256 + tile*2 + hh] = sred[hh*128 + ch];
  }
}

// --------------------------- down-projection ------------------------------
__global__ __launch_bounds__(256) void k_downproj(const float* __restrict__ schunk,
    const float* __restrict__ dw, const float* __restrict__ db, float* __restrict__ hout) {
  extern __shared__ unsigned char smem[];
  float* pooled = (float*)smem;        // [band][t]
  float* pe = pooled + 16384;          // [33][t]
  int b = blockIdx.x, tid = threadIdx.x;
  const float* sc = schunk + (size_t)b*128*256;
  for (int i = tid; i < 16384; i += 256) {
    int band = i >> 7, t = i & 127;
    float s = 0.f;
    int m0 = 2*t - 1;
    for (int u = 0; u < 4; ++u) { int m = m0 + u; if (m >= 0 && m < 256) s += sc[band*256 + m]; }
    pooled[i] = s*(1.f/512.f);
  }
  for (int i = tid; i < 33*128; i += 256) {
    int f = i >> 7, t = i & 127;
    float p = -PI_F + (float)t*(TPI_F/127.f);
    float val;
    if (f == 0) val = p;
    else { float arg = ldexpf(p, (f - 1) >> 1); val = (f & 1) ? sinf(arg) : cosf(arg); }
    pe[i] = val;
  }
  __syncthreads();
  for (int i = tid; i < 16384; i += 256) {
    int co = i >> 7, t = i & 127;
    float acc = db[co];
    for (int band = 0; band < 128; ++band) acc += pooled[band*128 + t]*dw[band*128 + co];
    for (int f = 0; f < 33; ++f)           acc += pe[f*128 + t]*dw[(128 + f)*128 + co];
    hout[((size_t)b*128 + co)*128 + t] = acc;
  }
}

// ------------------------- dilated residual block --------------------------
// y = lrelu(conv2(conv1(h)+b1)+b2 + h); per-channel stats to stats[wg][ch].
__global__ __launch_bounds__(256, 1) void k_convblock(const float* __restrict__ hsrc,
    float* __restrict__ hdst,
    const unsigned short* __restrict__ w1, const float* __restrict__ b1,
    const unsigned short* __restrict__ w2, const float* __restrict__ b2,
    float* __restrict__ stats, int dil) {
  extern __shared__ unsigned char smem[];
  unsigned short* w1l = (unsigned short*)smem;                    // 128x384
  unsigned short* w2l = (unsigned short*)(smem + 98304);          // 128x128
  unsigned short* hl  = (unsigned short*)(smem + 131072);         // 128x96 (82 valid)
  unsigned short* tl  = (unsigned short*)(smem + 131072 + 24576); // 128x64
  int tid = threadIdx.x, lane = tid & 31, wave = tid >> 5;
  int b = blockIdx.x >> 1, half = blockIdx.x & 1;
  const float* hb_ = hsrc + (size_t)b*16384;
  // async-stage w1 (96KB) and w2 (32KB) into LDS
  {
    unsigned l1 = (unsigned)(unsigned long long)(void*)w1l;
    const char* g1 = (const char*)w1;
    for (int i = tid; i < 6144; i += 256)
      async_copy16(l1 + (unsigned)i*16u, g1 + (size_t)i*16u);
    unsigned l2 = (unsigned)(unsigned long long)(void*)w2l;
    const char* g2 = (const char*)w2;
    for (int i = tid; i < 2048; i += 256)
      async_copy16(l2 + (unsigned)i*16u, g2 + (size_t)i*16u);
  }
  for (int i = tid; i < 128*96; i += 256) {
    int ci = i/96, j = i - ci*96;
    int t = half*64 - 9 + j;
    hl[i] = f2bf((j < 82 && t >= 0 && t < 128) ? hb_[ci*128 + t] : 0.f);
  }
  wait_async();
  __syncthreads();
  int mt = wave, kb = (lane >= 16) ? 16 : 0;
  v8f acc[4];
#pragma unroll
  for (int j = 0; j < 4; ++j)
#pragma unroll
    for (int r = 0; r < 8; ++r) acc[j][r] = 0.f;
  const unsigned short* arow = w1l + (mt*16 + (lane & 15))*384;
  for (int K0 = 0; K0 < 384; K0 += 32) {
    v16bf a = load_a(arow + K0, lane);
    v16bf bbf[4];
#pragma unroll
    for (int j = 0; j < 4; ++j) {
      int nloc = j*16 + (lane & 15);
#pragma unroll
      for (int v = 0; v < 8; ++v)
#pragma unroll
        for (int s2 = 0; s2 < 2; ++s2) {
          int k = K0 + kb + 2*v + s2;
          int ci = k/3, tap = k - ci*3;               // k = ci*3 + tap
          bbf[j][2*v + s2] = bfb(hl[ci*96 + nloc + 9 + (tap - 1)*dil]);
        }
    }
#pragma unroll
    for (int j = 0; j < 4; ++j) acc[j] = wmma_bf16(a, bbf[j], acc[j]);
    SCHED_GROUP(0x100, 66, 0);   // 2 A + 64 B ds reads first
    SCHED_GROUP(0x008, 4, 0);    // then 4 WMMAs
  }
  // t = conv1 + b1  -> LDS bf16
#pragma unroll
  for (int j = 0; j < 4; ++j)
#pragma unroll
    for (int r = 0; r < 8; ++r) {
      int ch = mt*16 + ((lane >= 16) ? r + 8 : r);
      int col = j*16 + (lane & 15);
      tl[ch*64 + col] = f2bf(acc[j][r] + b1[ch]);
    }
  __syncthreads();
  v8f acc2[4];
#pragma unroll
  for (int j = 0; j < 4; ++j)
#pragma unroll
    for (int r = 0; r < 8; ++r) acc2[j][r] = 0.f;
  const unsigned short* arow2 = w2l + (mt*16 + (lane & 15))*128;
  for (int K0 = 0; K0 < 128; K0 += 32) {
    v16bf a = load_a(arow2 + K0, lane);
    v16bf bbf[4];
#pragma unroll
    for (int j = 0; j < 4; ++j) {
      int nloc = j*16 + (lane & 15);
      const unsigned short* bp = tl + (K0 + kb)*64 + nloc;
#pragma unroll
      for (int v = 0; v < 8; ++v) { bbf[j][2*v] = bfb(bp[(2*v)*64]); bbf[j][2*v+1] = bfb(bp[(2*v+1)*64]); }
    }
#pragma unroll
    for (int j = 0; j < 4; ++j) acc2[j] = wmma_bf16(a, bbf[j], acc2[j]);
    SCHED_GROUP(0x100, 66, 0);
    SCHED_GROUP(0x008, 4, 0);
  }
  float* dst = hdst + (size_t)b*16384;
#pragma unroll
  for (int r = 0; r < 8; ++r) {
    int ch = mt*16 + ((lane >= 16) ? r + 8 : r);
    float bias = b2[ch];
    float s = 0.f, q = 0.f;
#pragma unroll
    for (int j = 0; j < 4; ++j) {
      int t = half*64 + j*16 + (lane & 15);
      float y = lrelu(acc2[j][r] + bias + hb_[ch*128 + t]);  // fp32 residual
      dst[ch*128 + t] = y;
      s += y; q += y*y;
    }
    for (int o = 8; o > 0; o >>= 1) { s += __shfl_xor(s, o, 32); q += __shfl_xor(q, o, 32); }
    if ((lane & 15) == 0) {                                  // single writer per (wg,ch)
      stats[blockIdx.x*256 + ch] = s;
      stats[blockIdx.x*256 + 128 + ch] = q;
    }
  }
}

__global__ void k_norm(float* __restrict__ h, const float* __restrict__ stats,
                       const float* __restrict__ g, const float* __restrict__ bb) {
  int idx = blockIdx.x*blockDim.x + threadIdx.x;
  if (idx >= 131072) return;
  int ch = (idx >> 7) & 127;
  float s = 0.f, q = 0.f;
  for (int w = 0; w < 16; ++w) { s += stats[w*256 + ch]; q += stats[w*256 + 128 + ch]; }
  float mu = s*(1.f/1024.f);
  float var = q*(1.f/1024.f) - mu*mu;
  h[idx] = (h[idx] - mu)*rsqrtf(var + 1e-5f)*g[ch] + bb[ch];
}

// ------------------- attention, top-k, events, context ---------------------
__global__ __launch_bounds__(256) void k_attention(const float* __restrict__ h,
    const float* __restrict__ attn_w, const float* __restrict__ attn_b,
    const float* __restrict__ proj_w, const float* __restrict__ proj_b,
    float* __restrict__ evout, float* __restrict__ ctxout) {
  extern __shared__ unsigned char smem[];
  float* hloc  = (float*)smem;        // [ch][t]
  float* hploc = hloc + 16384;        // [t][c]
  float* sv    = hploc + 16384;       // 128
  int*   si    = (int*)(sv + 128);    // 128
  __shared__ float red[1];
  int b = blockIdx.x, tid = threadIdx.x;
  const float* hb_ = h + (size_t)b*16384;
  for (int i = tid; i < 16384; i += 256) hloc[i] = hb_[i];
  __syncthreads();
  if (tid < 128) {
    float acc = attn_b[0];
    for (int ch = 0; ch < 128; ++ch) acc += hloc[ch*128 + tid]*attn_w[ch];
    sv[tid] = acc; si[tid] = tid;
  }
  __syncthreads();
  if (tid == 0) {
    float m = sv[0];
    for (int t = 1; t < 128; ++t) m = fmaxf(m, sv[t]);
    float s = 0.f;
    for (int t = 0; t < 128; ++t) { sv[t] = expf(sv[t] - m); s += sv[t]; }
    red[0] = s;
  }
  __syncthreads();
  if (tid < 128) sv[tid] *= 1.f/red[0];
  __syncthreads();
  for (int i = tid; i < 16384; i += 256) {
    int t = i >> 7, c = i & 127;
    float acc = proj_b[c];
    for (int ch = 0; ch < 128; ++ch) acc += hloc[ch*128 + t]*proj_w[ch*128 + c];
    hploc[i] = acc;
  }
  __syncthreads();
  if (tid < 128) {
    float s = 0.f;
    for (int t = 0; t < 128; ++t) s += hploc[t*128 + tid];
    ctxout[b*128 + tid] = s;
  }
  // bitonic ascending with top_k tie-break; top-64 read from the tail.
  for (int k = 2; k <= 128; k <<= 1)
    for (int j = k >> 1; j > 0; j >>= 1) {
      __syncthreads();
      if (tid < 128) {
        int ixj = tid ^ j;
        if (ixj > tid) {
          float v0 = sv[tid], v1 = sv[ixj]; int i0 = si[tid], i1 = si[ixj];
          bool up = ((tid & k) == 0);
          bool gt = (v0 > v1) || (v0 == v1 && i0 < i1);
          bool lt = (v0 < v1) || (v0 == v1 && i0 > i1);
          if (up ? gt : lt) { sv[tid] = v1; sv[ixj] = v0; si[tid] = i1; si[ixj] = i0; }
        }
      }
    }
  __syncthreads();
  for (int i = tid; i < 8192; i += 256) {
    int e = i >> 7, c = i & 127;
    int p = 127 - e;
    evout[(size_t)b*8192 + i] = hploc[si[p]*128 + c]*sv[p];
  }
}

// ------------------------ event-wise MLP stacks ----------------------------
// mode: 0 amp (|out|), 1 atom (argmax of 2048 logits), 2 pos (raw out)
__global__ __launch_bounds__(256) void k_stack_ev(const float* __restrict__ ev,
    const float* __restrict__ hw, const float* __restrict__ hb,
    const float* __restrict__ g,  const float* __restrict__ bt,
    const float* __restrict__ ow, const float* __restrict__ ob,
    int mode, float* __restrict__ sout, int* __restrict__ iout) {
  extern __shared__ unsigned char smem[];
  float* X = (float*)smem;      // 64x128
  float* Y = X + 8192;
  __shared__ float wred[8][2];
  __shared__ float bval[256]; __shared__ int bidx[256];
  int b = blockIdx.x, tid = threadIdx.x, lane = tid & 31, wv = tid >> 5;
  const float* evb = ev + (size_t)b*8192;
  for (int i = tid; i < 8192; i += 256) X[i] = evb[i];
  __syncthreads();
  for (int L = 0; L < 3; ++L) {
    const float* W = hw + L*16384;
    const float* bias = hb + L*128;
    float s = 0.f, q = 0.f;
    for (int i = tid; i < 8192; i += 256) {
      int e = i >> 7, c = i & 127;
      const float* xr = X + (e << 7);
      float acc = bias[c];
      for (int k = 0; k < 128; ++k) acc += xr[k]*W[k*128 + c];
      Y[i] = acc; s += acc; q += acc*acc;
    }
    for (int o = 16; o > 0; o >>= 1) { s += __shfl_down(s, o, 32); q += __shfl_down(q, o, 32); }
    if (lane == 0) { wred[wv][0] = s; wred[wv][1] = q; }
    __syncthreads();
    float st = 0.f, qt = 0.f;
    for (int w2 = 0; w2 < 8; ++w2) { st += wred[w2][0]; qt += wred[w2][1]; }
    float mu = st*(1.f/8192.f), var = qt*(1.f/8192.f) - mu*mu;
    float rstd = rsqrtf(var + 1e-5f);
    const float* gl = g + L*8192;
    const float* bl = bt + L*8192;
    for (int i = tid; i < 8192; i += 256)
      X[i] = lrelu((Y[i] - mu)*rstd*gl[i] + bl[i]);
    __syncthreads();
  }
  if (mode == 1) {
    int e = tid >> 2, part = tid & 3;
    const float* xr = X + (e << 7);
    float best = -1e30f; int bj = 0;
    for (int j = part; j < 2048; j += 4) {
      float acc = ob[j];
      for (int k = 0; k < 128; ++k) acc += xr[k]*ow[k*2048 + j];
      if (acc > best) { best = acc; bj = j; }
    }
    bval[tid] = best; bidx[tid] = bj;
    __syncthreads();
    if (part == 0) {
      float bv = bval[tid]; int bi = bidx[tid];
      for (int p2 = 1; p2 < 4; ++p2) {
        float v = bval[tid + p2]; int ii = bidx[tid + p2];
        if (v > bv || (v == bv && ii < bi)) { bv = v; bi = ii; }
      }
      iout[b*64 + e] = bi;
    }
  } else if (tid < 64) {
    const float* xr = X + (tid << 7);
    float acc = ob[0];
    for (int k = 0; k < 128; ++k) acc += xr[k]*ow[k];
    if (mode == 0) acc = fabsf(acc);
    sout[b*64 + tid] = acc;
  }
}

// ---------------------- context stacks (mix, rooms) ------------------------
__global__ __launch_bounds__(128) void k_ctx_stack(const float* __restrict__ ctx,
    const float* __restrict__ vm_hw, const float* __restrict__ vm_hb,
    const float* __restrict__ vm_g,  const float* __restrict__ vm_bt,
    const float* __restrict__ vm_ow, const float* __restrict__ vm_ob,
    const float* __restrict__ vr_hw, const float* __restrict__ vr_hb,
    const float* __restrict__ vr_g,  const float* __restrict__ vr_bt,
    const float* __restrict__ vr_ow, const float* __restrict__ vr_ob,
    float* __restrict__ mixo, float* __restrict__ rmo) {
  __shared__ float X[128], wred[4][2], outs[8];
  int b = blockIdx.x, tid = threadIdx.x, lane = tid & 31, wv = tid >> 5;
  for (int pass = 0; pass < 2; ++pass) {
    const float* hw = pass ? vr_hw : vm_hw;
    const float* hb = pass ? vr_hb : vm_hb;
    const float* gg = pass ? vr_g  : vm_g;
    const float* bt = pass ? vr_bt : vm_bt;
    const float* ow = pass ? vr_ow : vm_ow;
    const float* ob = pass ? vr_ob : vm_ob;
    X[tid] = ctx[b*128 + tid];
    __syncthreads();
    for (int L = 0; L < 3; ++L) {
      const float* W = hw + L*16384;
      float acc = hb[L*128 + tid];
      for (int k = 0; k < 128; ++k) acc += X[k]*W[k*128 + tid];
      __syncthreads();
      float s = acc, q = acc*acc;
      for (int o = 16; o > 0; o >>= 1) { s += __shfl_down(s, o, 32); q += __shfl_down(q, o, 32); }
      if (lane == 0) { wred[wv][0] = s; wred[wv][1] = q; }
      __syncthreads();
      float st = 0.f, qt = 0.f;
      for (int w2 = 0; w2 < 4; ++w2) { st += wred[w2][0]; qt += wred[w2][1]; }
      float mu = st*(1.f/128.f), var = qt*(1.f/128.f) - mu*mu;
      X[tid] = lrelu((acc - mu)*rsqrtf(var + 1e-5f)*gg[L*128 + tid] + bt[L*128 + tid]);
      __syncthreads();
    }
    if (pass == 0) {
      if (tid == 0) {
        float o = ob[0];
        for (int k = 0; k < 128; ++k) o += X[k]*ow[k];
        mixo[b] = 1.f/(1.f + expf(-o));
      }
    } else {
      if (tid < 8) {
        float o = ob[tid];
        for (int k = 0; k < 128; ++k) o += X[k]*ow[k*8 + tid];
        outs[tid] = o;
      }
      __syncthreads();
      if (tid == 0) {
        float m = outs[0];
        for (int j = 1; j < 8; ++j) m = fmaxf(m, outs[j]);
        float e[8], sum = 0.f;
        for (int j = 0; j < 8; ++j) { e[j] = expf(outs[j] - m); sum += e[j]; }
        for (int j = 0; j < 8; ++j) rmo[b*8 + j] = e[j]/sum;
      }
    }
    __syncthreads();
  }
}

__global__ void k_impulse(const float* __restrict__ rooms, const float* __restrict__ rm,
                          float* __restrict__ imp) {
  int i = blockIdx.x*blockDim.x + threadIdx.x;
  if (i >= 262144) return;
  int b = i >> 15, t = i & 32767;
  const float* r = rm + b*8;
  float s = 0.f;
  for (int q = 0; q < 8; ++q) s += r[q]*rooms[q*32768 + t];
  imp[i] = s;
}

// ------------------------------ FFT core -----------------------------------
// In-place radix-2 DIT complex FFT, N=32768, data in LDS.
__device__ void fftN(float2* z, float sign, int tid, int nthr) {
  for (int i = tid; i < 32768; i += nthr) {
    int j = (int)(__brev((unsigned)i) >> 17);
    if (j > i) { float2 t = z[i]; z[i] = z[j]; z[j] = t; }
  }
  __syncthreads();
  for (int s = 1; s <= 15; ++s) {
    int hh = 1 << (s - 1);
    float base = sign*TPI_F/(float)(1 << s);
    for (int idx = tid; idx < 16384; idx += nthr) {
      int pos = idx & (hh - 1);
      int i0 = ((idx >> (s - 1)) << s) + pos;
      int i1 = i0 + hh;
      float sn, cs; __sincosf(base*(float)pos, &sn, &cs);
      float2 u = z[i0], v = z[i1];
      float vr = v.x*cs - v.y*sn;
      float vi = v.x*sn + v.y*cs;
      z[i0] = make_float2(u.x + vr, u.y + vi);
      z[i1] = make_float2(u.x - vr, u.y - vi);
    }
    __syncthreads();
  }
}

// X[k] of the 65536-pt rfft from the packed 32768-pt spectrum Z (in LDS).
__device__ __forceinline__ float2 spec_at(const float2* z, int k) {
  int kk = k & 32767, mk = (32768 - k) & 32767;
  float2 Zk = z[kk], Zm = z[mk];
  float er = 0.5f*(Zk.x + Zm.x), ei = 0.5f*(Zk.y - Zm.y);
  float orr = 0.5f*(Zk.y + Zm.y), oi = -0.5f*(Zk.x - Zm.x);
  float sn, cs; __sincosf(-TPI_F*(float)k/65536.f, &sn, &cs);
  return make_float2(er + cs*orr - sn*oi, ei + cs*oi + sn*orr);
}

// ----------------------- event spectra (fft_shift) -------------------------
// WG = (batch, group-of-16 events); register-accumulated phased spectra.
__global__ __launch_bounds__(1024) void k_eventfft(const float* __restrict__ atoms,
    const float* __restrict__ anorm, const int* __restrict__ aidx,
    const float* __restrict__ amps, const float* __restrict__ posv,
    float* __restrict__ gspec) {
  extern __shared__ unsigned char smem[];
  float2* z = (float2*)smem;   // 32768 * 8B = 256KB
  int b = blockIdx.x >> 2, grp = blockIdx.x & 3;
  int tid = threadIdx.x;
  float accr[32], acci[32];
#pragma unroll
  for (int j = 0; j < 32; ++j) { accr[j] = 0.f; acci[j] = 0.f; }
  float lastr = 0.f, lasti = 0.f;
  for (int ei = 0; ei < 16; ++ei) {
    int e = grp*16 + ei;
    int row = aidx[b*64 + e];
    float sc = anorm[row]*amps[b*64 + e];
    double pv = (double)posv[b*64 + e];
    const float* arow = atoms + (size_t)row*2048;
    __syncthreads();
    for (int n = tid; n < 32768; n += 1024) {
      float2 v = make_float2(0.f, 0.f);
      if (n < 1024) { v.x = arow[2*n]*sc; v.y = arow[2*n + 1]*sc; }
      z[n] = v;
    }
    __syncthreads();
    fftN(z, -1.f, tid, 1024);
    // phase exp(-i*2*pi*k*pos*16384/32769), rotated recursively in double.
    double phc = -(2.0*PI_D)*16384.0/32769.0*pv;
    double c0, s0; sincos(phc*(double)tid, &s0, &c0);
    double dc, ds; sincos(phc*1024.0, &ds, &dc);
#pragma unroll
    for (int j = 0; j < 32; ++j) {
      int k = tid + j*1024;
      float2 X = spec_at(z, k);
      float cs = (float)c0, sn = (float)s0;
      accr[j] += X.x*cs - X.y*sn;
      acci[j] += X.x*sn + X.y*cs;
      double nc = c0*dc - s0*ds, nsv = c0*ds + s0*dc;
      c0 = nc; s0 = nsv;
    }
    if (tid == 0) {
      float2 X = spec_at(z, 32768);
      double sn, cs; sincos(phc*32768.0, &sn, &cs);
      lastr += X.x*(float)cs - X.y*(float)sn;
      lasti += X.x*(float)sn + X.y*(float)cs;
    }
  }
  float* gs = gspec + (size_t)blockIdx.x*32769*2;
#pragma unroll
  for (int j = 0; j < 32; ++j) { int k = tid + j*1024; gs[2*k] = accr[j]; gs[2*k + 1] = acci[j]; }
  if (tid == 0) { gs[2*32768] = lastr; gs[2*32768 + 1] = lasti; }
}

// -------------------------- dry = sum irfft --------------------------------
__global__ __launch_bounds__(1024) void k_dry(const float* __restrict__ gspec,
                                              float* __restrict__ dry) {
  extern __shared__ unsigned char smem[];
  float2* z = (float2*)smem;
  int b = blockIdx.x, tid = threadIdx.x;
  const float* g0 = gspec + (size_t)(b*4)*32769*2;
  for (int k = tid; k <= 16384; k += 1024) {
    float xr = 0.f, xi = 0.f, mr = 0.f, mi = 0.f;
    for (int q = 0; q < 4; ++q) {                // fixed-order group sum
      const float* gq = g0 + (size_t)q*32769*2;
      int km = 32768 - k;
      xr += gq[2*k]; xi += gq[2*k + 1];
      mr += gq[2*km]; mi += gq[2*km + 1];
    }
    if (k == 0) { xi = 0.f; mi = 0.f; }          // C2R drops DC/Nyquist imag
    float er = 0.5f*(xr + mr), ei2 = 0.5f*(xi - mi);
    float tr = 0.5f*(xr - mr), ti = 0.5f*(xi + mi);
    float sn, cs; __sincosf(TPI_F*(float)k/65536.f, &sn, &cs);
    float orr = tr*cs - ti*sn, oi = tr*sn + ti*cs;
    float2 Zk = make_float2(er - oi,  ei2 + orr);
    float2 Zm = make_float2(er + oi, -ei2 + orr);
    z[k & 32767] = Zk;
    if (k >= 1 && k < 16384) z[32768 - k] = Zm;
  }
  __syncthreads();
  fftN(z, +1.f, tid, 1024);
  float* d = dry + (size_t)b*32768;
  const float scale = 1.f/32768.f;
  for (int n = tid; n < 16384; n += 1024) {      // truncate to 32768 samples
    d[2*n]     = z[n].x*scale;
    d[2*n + 1] = z[n].y*scale;
  }
}

// ------------------- wet = irfft(rfft(dry)*rfft(imp)), final mix -----------
__global__ __launch_bounds__(1024) void k_wet(const float* __restrict__ dry,
    const float* __restrict__ imp, const float* __restrict__ mixv,
    float* __restrict__ wspec, float* __restrict__ outp) {
  extern __shared__ unsigned char smem[];
  float2* z = (float2*)smem;
  int b = blockIdx.x, tid = threadIdx.x;
  const float* d = dry + (size_t)b*32768;
  for (int n = tid; n < 32768; n += 1024) {
    float2 v = make_float2(0.f, 0.f);
    if (n < 16384) { v.x = d[2*n]; v.y = d[2*n + 1]; }
    z[n] = v;
  }
  __syncthreads();
  fftN(z, -1.f, tid, 1024);
  float* ws = wspec + (size_t)b*32769*2;
  for (int k = tid; k <= 32768; k += 1024) {
    float2 X = spec_at(z, k);
    ws[2*k] = X.x; ws[2*k + 1] = X.y;
  }
  __threadfence_block();
  __syncthreads();
  const float* ip = imp + (size_t)b*32768;
  for (int n = tid; n < 32768; n += 1024) {
    float2 v = make_float2(0.f, 0.f);
    if (n < 16384) { v.x = ip[2*n]; v.y = ip[2*n + 1]; }
    z[n] = v;
  }
  __syncthreads();
  fftN(z, -1.f, tid, 1024);
  for (int k = tid; k <= 16384; k += 1024) {
    int kk = k & 32767, mk = (32768 - k) & 32767;
    float2 Zk = z[kk], Zm = z[mk];
    float er = 0.5f*(Zk.x + Zm.x), ei2 = 0.5f*(Zk.y - Zm.y);
    float o0 = 0.5f*(Zk.y + Zm.y), o1 = -0.5f*(Zk.x - Zm.x);
    float sn, cs; __sincosf(-TPI_F*(float)k/65536.f, &sn, &cs);
    float wor = cs*o0 - sn*o1, woi = cs*o1 + sn*o0;
    float2 Xik = make_float2(er + wor,  ei2 + woi);      // Ximp[k]
    float2 Xim = make_float2(er - wor, -(ei2 - woi));    // Ximp[N-k] = conj(E - W*O)
    float2 Xdk = make_float2(ws[2*k], ws[2*k + 1]);
    float2 Xdm = make_float2(ws[2*(32768 - k)], ws[2*(32768 - k) + 1]);
    float2 Yk = cmulf(Xdk, Xik);
    float2 Ym = cmulf(Xdm, Xim);
    if (k == 0) { Yk.y = 0.f; Ym.y = 0.f; }              // DC/Nyquist real
    float e2r = 0.5f*(Yk.x + Ym.x), e2i = 0.5f*(Yk.y - Ym.y);
    float t2r = 0.5f*(Yk.x - Ym.x), t2i = 0.5f*(Yk.y + Ym.y);
    float o2r = cs*t2r + sn*t2i;                         // conj(W)*T
    float o2i = cs*t2i - sn*t2r;
    float2 Znk = make_float2(e2r - o2i,  e2i + o2r);
    float2 Znm = make_float2(e2r + o2i, -e2i + o2r);
    z[kk] = Znk;
    if (k >= 1 && k < 16384) z[mk] = Znm;
  }
  __syncthreads();
  fftN(z, +1.f, tid, 1024);
  float mx = mixv[b];
  float* ob_ = outp + (size_t)b*32768;
  const float scale = 1.f/32768.f;
  for (int n = tid; n < 16384; n += 1024) {
    float w0 = z[n].x*scale, w1 = z[n].y*scale;
    ob_[2*n]     = d[2*n]*mx     + w0*(1.f - mx);
    ob_[2*n + 1] = d[2*n + 1]*mx + w1*(1.f - mx);
  }
}

// ------------------------------- launcher ----------------------------------
extern "C" void kernel_launch(void* const* d_in, const int* in_sizes, int n_in,
                              void* d_out, int out_size, void* d_ws, size_t ws_size,
                              hipStream_t stream) {
  (void)in_sizes; (void)n_in; (void)out_size; (void)ws_size;
  const float* x      = (const float*)d_in[0];
  const float* fb     = (const float*)d_in[1];
  const float* down_w = (const float*)d_in[2];
  const float* down_b = (const float*)d_in[3];
  const float* c1w    = (const float*)d_in[4];
  const float* c1b    = (const float*)d_in[5];
  const float* c2w    = (const float*)d_in[6];
  const float* c2b    = (const float*)d_in[7];
  const float* bng    = (const float*)d_in[8];
  const float* bnb    = (const float*)d_in[9];
  const float* proj_w = (const float*)d_in[10];
  const float* proj_b = (const float*)d_in[11];
  const float* attn_w = (const float*)d_in[12];
  const float* attn_b = (const float*)d_in[13];
  const float* atoms  = (const float*)d_in[14];
  const float* amp_hw = (const float*)d_in[15];
  const float* amp_hb = (const float*)d_in[16];
  const float* amp_g  = (const float*)d_in[17];
  const float* amp_bt = (const float*)d_in[18];
  const float* amp_ow = (const float*)d_in[19];
  const float* amp_ob = (const float*)d_in[20];
  const float* atom_hw = (const float*)d_in[21];
  const float* atom_hb = (const float*)d_in[22];
  const float* atom_g  = (const float*)d_in[23];
  const float* atom_bt = (const float*)d_in[24];
  const float* atom_ow = (const float*)d_in[25];
  const float* atom_ob = (const float*)d_in[26];
  const float* pos_hw = (const float*)d_in[27];
  const float* pos_hb = (const float*)d_in[28];
  const float* pos_g  = (const float*)d_in[29];
  const float* pos_bt = (const float*)d_in[30];
  const float* pos_ow = (const float*)d_in[31];
  const float* pos_ob = (const float*)d_in[32];
  const float* vm_hw = (const float*)d_in[33];
  const float* vm_hb = (const float*)d_in[34];
  const float* vm_g  = (const float*)d_in[35];
  const float* vm_bt = (const float*)d_in[36];
  const float* vm_ow = (const float*)d_in[37];
  const float* vm_ob = (const float*)d_in[38];
  const float* vr_hw = (const float*)d_in[39];
  const float* vr_hb = (const float*)d_in[40];
  const float* vr_g  = (const float*)d_in[41];
  const float* vr_bt = (const float*)d_in[42];
  const float* vr_ow = (const float*)d_in[43];
  const float* vr_ob = (const float*)d_in[44];
  const float* rooms = (const float*)d_in[45];

  char* base = (char*)d_ws;
  size_t off = 0;
  auto take = [&](size_t bytes) -> char* {
    char* p = base + off;
    off += (bytes + 255) & ~(size_t)255;
    return p;
  };
  unsigned short* fbbf = (unsigned short*)take(65536*2);
  unsigned short* w1bf = (unsigned short*)take((size_t)344064*2);
  unsigned short* w2bf = (unsigned short*)take((size_t)114688*2);
  float* schunk = (float*)take((size_t)262144*4);
  float* hA     = (float*)take((size_t)131072*4);
  float* hB     = (float*)take((size_t)131072*4);
  float* stats  = (float*)take((size_t)7*16*256*4);
  float* evb    = (float*)take((size_t)65536*4);
  float* ctx    = (float*)take(1024*4);
  float* ampb   = (float*)take(512*4);
  float* posb   = (float*)take(512*4);
  int*   aidx   = (int*)take(512*4);
  float* anrm   = (float*)take(2048*4);
  float* mixb   = (float*)take(8*4);
  float* rmb    = (float*)take(64*4);
  float* impb   = (float*)take((size_t)262144*4);
  float* dryb   = (float*)take((size_t)262144*4);
  float* gspec  = (float*)take((size_t)32*32769*2*4);
  float* wspec  = (float*)take((size_t)8*32769*2*4);

  // prep: bf16 weight copies + atom norms
  k_cvt<<<(65536 + 255)/256, 256, 0, stream>>>(fb, fbbf, 65536);
  k_cvt<<<(344064 + 255)/256, 256, 0, stream>>>(c1w, w1bf, 344064);
  k_cvt<<<(114688 + 255)/256, 256, 0, stream>>>(c2w, w2bf, 114688);
  k_atomnorm<<<2048, 256, 0, stream>>>(atoms, anrm);

  // filterbank (WMMA) + pooling chunk sums
  k_filterbank<<<1024, 512, 131072 + 2048 + 1024, stream>>>(x, fbbf, schunk);
  // down projection with positional encodings
  k_downproj<<<8, 256, (16384 + 33*128)*4, stream>>>(schunk, down_w, down_b, hA);

  // 7 dilated residual blocks (WMMA) + cross-batch norm
  const int dil[7] = {1, 3, 9, 1, 3, 9, 1};
  float* cur = hA; float* oth = hB;
  for (int i = 0; i < 7; ++i) {
    k_convblock<<<16, 256, 98304 + 32768 + 24576 + 16384, stream>>>(
        cur, oth, w1bf + i*49152, c1b + i*128, w2bf + i*16384, c2b + i*128,
        stats + i*4096, dil[i]);
    k_norm<<<512, 256, 0, stream>>>(oth, stats + i*4096, bng + i*128, bnb + i*128);
    float* t = cur; cur = oth; oth = t;
  }

  // attention + softmax + top-64 + events + context
  k_attention<<<8, 256, (16384 + 16384)*4 + 128*4 + 128*4, stream>>>(
      cur, attn_w, attn_b, proj_w, proj_b, evb, ctx);

  // per-event MLP stacks
  k_stack_ev<<<8, 256, 65536, stream>>>(evb, amp_hw, amp_hb, amp_g, amp_bt, amp_ow, amp_ob, 0, ampb, aidx);
  k_stack_ev<<<8, 256, 65536, stream>>>(evb, atom_hw, atom_hb, atom_g, atom_bt, atom_ow, atom_ob, 1, ampb, aidx);
  k_stack_ev<<<8, 256, 65536, stream>>>(evb, pos_hw, pos_hb, pos_g, pos_bt, pos_ow, pos_ob, 2, posb, aidx);

  // context stacks (mix sigmoid, room softmax) + room impulse
  k_ctx_stack<<<8, 128, 0, stream>>>(ctx, vm_hw, vm_hb, vm_g, vm_bt, vm_ow, vm_ob,
                                     vr_hw, vr_hb, vr_g, vr_bt, vr_ow, vr_ob, mixb, rmb);
  k_impulse<<<1024, 256, 0, stream>>>(rooms, rmb, impb);

  // event spectra (fft_shift), dry irfft, wet convolution + final mix
  k_eventfft<<<32, 1024, 262144, stream>>>(atoms, anrm, aidx, ampb, posb, gspec);
  k_dry<<<8, 1024, 262144, stream>>>(gspec, dryb);
  k_wet<<<8, 1024, 262144, stream>>>(dryb, impb, mixb, wspec, (float*)d_out);
}
